// LMNPTransformerXL_70970039599574
// MI455X (gfx1250) — matrix operand order repaired and checked by reading, not compiled, not run on hardware
//
#include <hip/hip_runtime.h>
#include <hip/hip_bf16.h>

// ---------------- problem dims (fixed by reference) ----------------
constexpr int L = 8, NH = 16, DH = 64, DM = 1024, DI = 4096;
constexpr int BS = 8, XLEN = 512, MLEN = 512;
constexpr int SEQ = XLEN + MLEN;       // 1024
constexpr int QKV = 3 * NH * DH;       // 3072
constexpr int HD  = NH * DH;           // 1024

// ---------------- WMMA types (gfx1250, wave32) ----------------
typedef __attribute__((ext_vector_type(16))) __bf16 bf16x16;
typedef __attribute__((ext_vector_type(8)))  float  f32x8;

union FragBF { unsigned int u[8]; uint4 q[2]; bf16x16 bf; };
union FragF  { float f[8]; f32x8 v; };

// f32 pair -> packed bf16 dword, round-half-up, single v_perm_b32 for the pack
__device__ __forceinline__ unsigned pack2(float lo, float hi) {
  unsigned a = __float_as_uint(lo) + 0x8000u;
  unsigned b = __float_as_uint(hi) + 0x8000u;
  return __builtin_amdgcn_perm(b, a, 0x07060302u);  // {b[31:16], a[31:16]}
}

__device__ __forceinline__ float wredSum(float v) {
#pragma unroll
  for (int o = 16; o > 0; o >>= 1) v += __shfl_xor(v, o, 32);
  return v;
}
__device__ __forceinline__ float wredMax(float v) {
#pragma unroll
  for (int o = 16; o > 0; o >>= 1) v = fmaxf(v, __shfl_xor(v, o, 32));
  return v;
}

// ---------------- generic batched bf16-WMMA GEMM ----------------
// Block tile 128x64, K step 32, double-buffered LDS. 256 threads = 8 waves
// (4x2), wave tile 32x32 -> 4 x v_wmma_f32_16x16x32_bf16 per K step.
// 80B LDS row pitch: every fragment half is one 16B ds_load_b128.
constexpr int BMt = 128, BNt = 64, BKt = 32, PIT = 40;  // PIT in ushorts

template <int TRANSB, int SHIFTADD>
__global__ __launch_bounds__(256) void wmma_gemm_kernel(
    const float* __restrict__ A, long long sA1, long long sA2, int lda,
    const float* __restrict__ B, long long sB1, long long sB2, int ldb,
    float*       __restrict__ C, long long sC1, long long sC2, int ldc,
    int M, int N, int K, int batch2,
    const float* __restrict__ bias,
    const float* __restrict__ res, int ldres, int relu)
{
  const int z  = blockIdx.z;
  const int b1 = z / batch2, b2 = z % batch2;
  A += b1 * sA1 + b2 * sA2;
  B += b1 * sB1 + b2 * sB2;
  C += b1 * sC1 + b2 * sC2;

  const int m0 = blockIdx.y * BMt;
  const int n0 = blockIdx.x * BNt;
  const int tid  = threadIdx.x;
  const int wave = tid >> 5;
  const int lane = tid & 31;
  const int wr  = wave >> 1;    // 0..3 : 32-row slab
  const int wc  = wave & 1;     // 0..1 : 32-col slab
  const int g   = lane >> 4;    // half-wave (ISA K-group)
  const int l15 = lane & 15;

  __shared__ alignas(16) unsigned short As[2][BMt][PIT];  // [m][k] bf16 bits
  __shared__ alignas(16) unsigned short Bs[2][BNt][PIT];  // [n][k] bf16 bits
  constexpr int ABUF = BMt * PIT;  // elements per A buffer
  constexpr int BBUF = BNt * PIT;  // elements per B buffer

  // ---- precompute staging pointers (advance by BK per tile) ----
  const float* aPtr[4]; unsigned short* aSt[4];
#pragma unroll
  for (int rr = 0; rr < 4; ++rr) {
    int c = rr * 256 + tid;
    int mm = c >> 3, kq = (c & 7) << 2;
    aPtr[rr] = A + (long long)(m0 + mm) * lda + kq;
    aSt[rr]  = &As[0][mm][kq];
  }
  const float* bPtr[2]; unsigned short* bSt[4];
  if (TRANSB) {
#pragma unroll
    for (int rr = 0; rr < 2; ++rr) {
      int c = rr * 256 + tid;
      int nn = c >> 3, kq = (c & 7) << 2;
      bPtr[rr] = B + (long long)(n0 + nn) * ldb + kq;
      bSt[rr]  = &Bs[0][nn][kq];
    }
  } else {
    int kk = (tid >> 4) << 1;        // 0,2,..,30
    int nq = (tid & 15) << 2;        // 0,4,..,60
    bPtr[0] = B + (long long)kk * ldb + n0 + nq;
    bPtr[1] = bPtr[0] + ldb;
    bSt[0] = &Bs[0][nq + 0][kk]; bSt[1] = &Bs[0][nq + 1][kk];
    bSt[2] = &Bs[0][nq + 2][kk]; bSt[3] = &Bs[0][nq + 3][kk];
  }

  float4 aR[4], bR[2];
  auto loadNext = [&]() {   // issue next tile's global loads, advance pointers
#pragma unroll
    for (int rr = 0; rr < 4; ++rr) { aR[rr] = *(const float4*)aPtr[rr]; aPtr[rr] += BKt; }
    if (TRANSB) {
#pragma unroll
      for (int rr = 0; rr < 2; ++rr) { bR[rr] = *(const float4*)bPtr[rr]; bPtr[rr] += BKt; }
    } else {
      bR[0] = *(const float4*)bPtr[0]; bPtr[0] += (long long)BKt * ldb;
      bR[1] = *(const float4*)bPtr[1]; bPtr[1] += (long long)BKt * ldb;
    }
  };
  auto storeTile = [&](int buf) {  // convert + store staged regs into LDS buffer
    const int ao = buf * ABUF, bo = buf * BBUF;
#pragma unroll
    for (int rr = 0; rr < 4; ++rr)
      *(uint2*)(aSt[rr] + ao) = make_uint2(pack2(aR[rr].x, aR[rr].y), pack2(aR[rr].z, aR[rr].w));
    if (TRANSB) {
#pragma unroll
      for (int rr = 0; rr < 2; ++rr)
        *(uint2*)(bSt[rr] + bo) = make_uint2(pack2(bR[rr].x, bR[rr].y), pack2(bR[rr].z, bR[rr].w));
    } else {
      *(unsigned*)(bSt[0] + bo) = pack2(bR[0].x, bR[1].x);
      *(unsigned*)(bSt[1] + bo) = pack2(bR[0].y, bR[1].y);
      *(unsigned*)(bSt[2] + bo) = pack2(bR[0].z, bR[1].z);
      *(unsigned*)(bSt[3] + bo) = pack2(bR[0].w, bR[1].w);
    }
  };

  f32x8 acc00 = {}, acc01 = {}, acc10 = {}, acc11 = {};

  // prologue: stage tile 0 into buffer 0
  loadNext();
  storeTile(0);
  __syncthreads();

  const int KT = K / BKt;
#pragma unroll 2
  for (int kt = 0; kt < KT; ++kt) {
    const int cur = kt & 1;
    const bool more = (kt + 1) < KT;
    if (more) {
      loadNext();                                   // overlap with WMMAs below
      if (kt + 2 < KT) __builtin_prefetch(aPtr[0] + BKt, 0, 3);
    }

    // ---- gather fragments (ISA 7.12.2 bf16 layouts; 2x ds_load_b128 each) ----
    const unsigned short (*Ac)[PIT] = As[cur];
    const unsigned short (*Bc)[PIT] = Bs[cur];
    FragBF a0, a1, b0, b1;
    {
      const unsigned short* Ar0 = Ac[wr * 32 + l15];
      const unsigned short* Ar1 = Ac[wr * 32 + 16 + l15];
      a0.q[0] = *(const uint4*)(Ar0 + 8 * g);        // K = 8g..8g+7
      a0.q[1] = *(const uint4*)(Ar0 + 16 + 8 * g);   // K = 16+8g..
      a1.q[0] = *(const uint4*)(Ar1 + 8 * g);
      a1.q[1] = *(const uint4*)(Ar1 + 16 + 8 * g);
      const unsigned short* Br0 = Bc[wc * 32 + l15];
      const unsigned short* Br1 = Bc[wc * 32 + 16 + l15];
      b0.q[0] = *(const uint4*)(Br0 + 8 * g);        // K = 8g..8g+7
      b0.q[1] = *(const uint4*)(Br0 + 16 + 8 * g);   // K = 16+8g..
      b1.q[0] = *(const uint4*)(Br1 + 8 * g);
      b1.q[1] = *(const uint4*)(Br1 + 16 + 8 * g);
    }
    acc00 = __builtin_amdgcn_wmma_f32_16x16x32_bf16(false, a0.bf, false, b0.bf, (short)0, acc00, false, false);
    acc01 = __builtin_amdgcn_wmma_f32_16x16x32_bf16(false, a0.bf, false, b1.bf, (short)0, acc01, false, false);
    acc10 = __builtin_amdgcn_wmma_f32_16x16x32_bf16(false, a1.bf, false, b0.bf, (short)0, acc10, false, false);
    acc11 = __builtin_amdgcn_wmma_f32_16x16x32_bf16(false, a1.bf, false, b1.bf, (short)0, acc11, false, false);

    if (more) storeTile(cur ^ 1);                    // fill the other buffer
    __syncthreads();
  }

  // ---- epilogue: C/D layout -> element e is row (e + 8g), col = lane&15 ----
#pragma unroll
  for (int mi = 0; mi < 2; ++mi) {
#pragma unroll
    for (int ni = 0; ni < 2; ++ni) {
      FragF fr;
      fr.v = (mi == 0) ? (ni == 0 ? acc00 : acc01) : (ni == 0 ? acc10 : acc11);
      const int n = n0 + wc * 32 + ni * 16 + l15;
#pragma unroll
      for (int e = 0; e < 8; ++e) {
        const int m = m0 + wr * 32 + mi * 16 + 8 * g + e;
        float val = fr.f[e];
        if (!SHIFTADD) {
          if (bias) val += bias[n];
          if (res)  val += res[(long long)m * ldres + n];
          if (relu) val = fmaxf(val, 0.0f);
          C[(long long)m * ldc + n] = val;
        } else {
          // TransformerXL _line_shift scatter: raw (m,n) -> flat f = m*(N+1)+n+1,
          // out = ((f-M)/N, (f-M)%N) if f >= M. Injective -> race-free +=.
          long long f_ = (long long)m * (N + 1) + n + 1;
          if (f_ >= M) {
            long long off = f_ - (long long)M;
            int io = (int)(off / N), jo = (int)(off % N);
            if (io < M) C[(long long)io * ldc + jo] += val;
          }
        }
      }
    }
  }
}

// ---------------- elementwise / reduction kernels ----------------
__global__ __launch_bounds__(256) void embed_kernel(const int* __restrict__ x,
                                                    const float* __restrict__ emb,
                                                    float* __restrict__ h) {
  long long idx = (long long)blockIdx.x * 256 + threadIdx.x;
  if (idx >= (long long)BS * XLEN * DM) return;
  int d = (int)(idx % DM);
  long long t = idx / DM;
  h[idx] = emb[(long long)x[t] * DM + d];
}

__global__ __launch_bounds__(256) void posenc_kernel(float* __restrict__ r) {
  long long idx = (long long)blockIdx.x * 256 + threadIdx.x;
  if (idx >= (long long)SEQ * DM) return;
  int pos = (int)(idx / DM), d = (int)(idx % DM);
  float p = (float)(SEQ - 1 - pos);
  int j = (d < DM / 2) ? d : d - DM / 2;
  float freq = __expf(-(float)(2 * j) * (9.210340371976184f / (float)DM));
  float a = p * freq;
  r[idx] = (d < DM / 2) ? sinf(a) : cosf(a);
}

__global__ __launch_bounds__(256) void concat_kernel(const float* __restrict__ memL,
                                                     const float* __restrict__ h,
                                                     float* __restrict__ ctx) {
  long long idx = (long long)blockIdx.x * 256 + threadIdx.x;
  if (idx >= (long long)BS * SEQ * DM) return;
  int d = (int)(idx % DM);
  long long t = idx / DM;
  int s = (int)(t % SEQ);
  int b = (int)(t / SEQ);
  ctx[idx] = (s < MLEN) ? memL[((long long)b * MLEN + s) * DM + d]
                        : h[((long long)b * XLEN + (s - MLEN)) * DM + d];
}

__global__ __launch_bounds__(256) void quqv_kernel(const float* __restrict__ qkv,
                                                   const float* __restrict__ u,
                                                   const float* __restrict__ v,
                                                   float* __restrict__ qu,
                                                   float* __restrict__ qv) {
  long long idx = (long long)blockIdx.x * 256 + threadIdx.x;
  if (idx >= (long long)BS * NH * XLEN * DH) return;
  int d = (int)(idx & (DH - 1));
  long long t = idx >> 6;                 // /DH
  int i = (int)(t % XLEN); t /= XLEN;
  int h = (int)(t % NH);
  int b = (int)(t / NH);
  float q = qkv[((long long)b * SEQ + MLEN + i) * QKV + h * DH + d];
  qu[idx] = q + u[h * DH + d];
  qv[idx] = q + v[h * DH + d];
}

__global__ __launch_bounds__(256) void softmax_kernel(float* __restrict__ score, float scale) {
  const int q = blockIdx.x;
  float* s = score + ((long long)blockIdx.y * XLEN + q) * SEQ;
  const int limit = MLEN + q;             // keys j <= MLEN + q are visible
  const int tid = threadIdx.x;
  const int wid = tid >> 5;
  __shared__ float red[8];
  float v[4]; float mx = -3.0e38f;
#pragma unroll
  for (int r = 0; r < 4; ++r) {
    int j = tid + 256 * r;
    float t = (j <= limit) ? s[j] * scale : -1.0e30f;
    v[r] = t; mx = fmaxf(mx, t);
  }
  mx = wredMax(mx);
  if ((tid & 31) == 0) red[wid] = mx;
  __syncthreads();
  float bmax = red[0];
#pragma unroll
  for (int i = 1; i < 8; ++i) bmax = fmaxf(bmax, red[i]);
  __syncthreads();
  float se = 0.0f;
#pragma unroll
  for (int r = 0; r < 4; ++r) { v[r] = __expf(v[r] - bmax); se += v[r]; }
  se = wredSum(se);
  if ((tid & 31) == 0) red[wid] = se;
  __syncthreads();
  float tot = 0.0f;
#pragma unroll
  for (int i = 0; i < 8; ++i) tot += red[i];
  float inv = 1.0f / tot;
#pragma unroll
  for (int r = 0; r < 4; ++r) s[tid + 256 * r] = v[r] * inv;
}

__global__ __launch_bounds__(256) void ln_kernel(const float* __restrict__ x,
                                                 const float* __restrict__ gg,
                                                 const float* __restrict__ bb,
                                                 float* __restrict__ out) {
  const float* xr = x + (long long)blockIdx.x * DM;
  float* yr = out + (long long)blockIdx.x * DM;
  const int tid = threadIdx.x;
  const int wid = tid >> 5;
  __shared__ float red[8];
  float v[4]; float s = 0.0f;
#pragma unroll
  for (int r = 0; r < 4; ++r) { v[r] = xr[tid + 256 * r]; s += v[r]; }
  s = wredSum(s);
  if ((tid & 31) == 0) red[wid] = s;
  __syncthreads();
  float tot = 0.0f;
#pragma unroll
  for (int i = 0; i < 8; ++i) tot += red[i];
  const float mean = tot * (1.0f / DM);
  __syncthreads();
  float s2 = 0.0f;
#pragma unroll
  for (int r = 0; r < 4; ++r) { float d = v[r] - mean; s2 += d * d; }
  s2 = wredSum(s2);
  if ((tid & 31) == 0) red[wid] = s2;
  __syncthreads();
  float tot2 = 0.0f;
#pragma unroll
  for (int i = 0; i < 8; ++i) tot2 += red[i];
  const float rstd = rsqrtf(tot2 * (1.0f / DM) + 1e-5f);
#pragma unroll
  for (int r = 0; r < 4; ++r) {
    int j = tid + 256 * r;
    yr[j] = (v[r] - mean) * rstd * gg[j] + bb[j];
  }
}

// ---------------- host orchestration ----------------
extern "C" void kernel_launch(void* const* d_in, const int* in_sizes, int n_in,
                              void* d_out, int out_size, void* d_ws, size_t ws_size,
                              hipStream_t stream) {
  (void)in_sizes; (void)n_in; (void)out_size; (void)ws_size;
  const int*   x     = (const int*)  d_in[0];
  const float* mem   = (const float*)d_in[1];
  const float* emb   = (const float*)d_in[2];
  const float* u_    = (const float*)d_in[3];
  const float* v_    = (const float*)d_in[4];
  const float* qkv_w = (const float*)d_in[5];
  const float* r_w   = (const float*)d_in[6];
  const float* out_w = (const float*)d_in[7];
  const float* ln1_g = (const float*)d_in[8];
  const float* ln1_b = (const float*)d_in[9];
  const float* ff1_w = (const float*)d_in[10];
  const float* ff1_b = (const float*)d_in[11];
  const float* ff2_w = (const float*)d_in[12];
  const float* ff2_b = (const float*)d_in[13];
  const float* ln2_g = (const float*)d_in[14];
  const float* ln2_b = (const float*)d_in[15];

  // workspace carve-up (256B aligned)
  char* wsp = (char*)d_ws;
  auto carve = [&](size_t bytes) -> float* {
    float* p = (float*)wsp;
    wsp += (bytes + 255) & ~(size_t)255;
    return p;
  };
  float* hbuf  = carve((size_t)BS * XLEN * DM * 4);       // 16 MB
  float* tmp   = carve((size_t)BS * XLEN * DM * 4);       // 16 MB
  float* ctx   = carve((size_t)BS * SEQ * DM * 4);        // 32 MB
  float* qkvb  = carve((size_t)BS * SEQ * QKV * 4);       // 96 MB
  float* rbuf  = carve((size_t)SEQ * DM * 4);             // 4 MB
  float* wkr   = carve((size_t)SEQ * HD * 4);             // 4 MB
  float* qu    = carve((size_t)BS * NH * XLEN * DH * 4);  // 16 MB
  float* qv    = carve((size_t)BS * NH * XLEN * DH * 4);  // 16 MB
  float* score = carve((size_t)BS * NH * XLEN * SEQ * 4); // 256 MB
  float* avb   = carve((size_t)BS * XLEN * HD * 4);       // 16 MB
  float* ffb   = carve((size_t)BS * XLEN * DI * 4);       // 64 MB

  auto gemm = [&](const float* A, long long sA1, long long sA2, int lda,
                  const float* B, long long sB1, long long sB2, int ldb, int transB,
                  float* C, long long sC1, long long sC2, int ldc,
                  int M, int N, int K, int b1, int b2,
                  const float* bias, const float* res, int ldres,
                  int relu, int shiftadd) {
    dim3 grid(N / BNt, M / BMt, b1 * b2);
    if (shiftadd) {
      if (transB) wmma_gemm_kernel<1, 1><<<grid, 256, 0, stream>>>(
          A, sA1, sA2, lda, B, sB1, sB2, ldb, C, sC1, sC2, ldc,
          M, N, K, b2, bias, res, ldres, relu);
      else        wmma_gemm_kernel<0, 1><<<grid, 256, 0, stream>>>(
          A, sA1, sA2, lda, B, sB1, sB2, ldb, C, sC1, sC2, ldc,
          M, N, K, b2, bias, res, ldres, relu);
    } else {
      if (transB) wmma_gemm_kernel<1, 0><<<grid, 256, 0, stream>>>(
          A, sA1, sA2, lda, B, sB1, sB2, ldb, C, sC1, sC2, ldc,
          M, N, K, b2, bias, res, ldres, relu);
      else        wmma_gemm_kernel<0, 0><<<grid, 256, 0, stream>>>(
          A, sA1, sA2, lda, B, sB1, sB2, ldb, C, sC1, sC2, ldc,
          M, N, K, b2, bias, res, ldres, relu);
    }
  };

  // encoder + positional encoding
  embed_kernel<<<(BS * XLEN * DM + 255) / 256, 256, 0, stream>>>(x, emb, hbuf);
  posenc_kernel<<<(SEQ * DM + 255) / 256, 256, 0, stream>>>(rbuf);

  const float scale = 0.125f;  // 1/sqrt(DH)

  for (int i = 0; i < L; ++i) {
    const float* memL   = mem   + (long long)i * BS * MLEN * DM;
    const float* qkvw_i = qkv_w + (long long)i * DM * QKV;
    const float* rw_i   = r_w   + (long long)i * DM * HD;
    const float* ow_i   = out_w + (long long)i * HD * DM;
    const float* f1w_i  = ff1_w + (long long)i * DM * DI;
    const float* f1b_i  = ff1_b + (long long)i * DI;
    const float* f2w_i  = ff2_w + (long long)i * DI * DM;
    const float* f2b_i  = ff2_b + (long long)i * DM;

    // ctx = concat(mem[i], h)
    concat_kernel<<<(BS * SEQ * DM + 255) / 256, 256, 0, stream>>>(memL, hbuf, ctx);

    // qkv = ctx @ qkv_w[i]           (8192 x 3072 x 1024)
    gemm(ctx, 0, 0, DM, qkvw_i, 0, 0, QKV, 0, qkvb, 0, 0, QKV,
         BS * SEQ, QKV, DM, 1, 1, nullptr, nullptr, 0, 0, 0);

    // wkr = r @ r_w[i]               (1024 x 1024 x 1024)
    gemm(rbuf, 0, 0, DM, rw_i, 0, 0, HD, 0, wkr, 0, 0, HD,
         SEQ, HD, DM, 1, 1, nullptr, nullptr, 0, 0, 0);

    // qu = q + u, qv = q + v  (per-head layout [b][h][i][d])
    quqv_kernel<<<(BS * NH * XLEN * DH + 255) / 256, 256, 0, stream>>>(qkvb, u_, v_, qu, qv);

    // AC = (q+u) @ k^T  per (b,h)    (512 x 1024 x 64) x128
    gemm(qu, (long long)NH * XLEN * DH, (long long)XLEN * DH, DH,
         qkvb + HD, (long long)SEQ * QKV, DH, QKV, 1,
         score, (long long)NH * XLEN * SEQ, (long long)XLEN * SEQ, SEQ,
         XLEN, SEQ, DH, BS, NH, nullptr, nullptr, 0, 0, 0);

    // score += line_shift((q+v) @ wkr^T)  — fused scatter-add epilogue
    gemm(qv, (long long)NH * XLEN * DH, (long long)XLEN * DH, DH,
         wkr, 0, DH, HD, 1,
         score, (long long)NH * XLEN * SEQ, (long long)XLEN * SEQ, SEQ,
         XLEN, SEQ, DH, BS, NH, nullptr, nullptr, 0, 0, /*shiftadd*/1);

    // masked softmax over keys (scale applied here)
    softmax_kernel<<<dim3(XLEN, BS * NH), 256, 0, stream>>>(score, scale);

    // av[b,q,h,d] = p @ w  per (b,h) (512 x 64 x 1024) x128
    gemm(score, (long long)NH * XLEN * SEQ, (long long)XLEN * SEQ, SEQ,
         qkvb + 2 * HD, (long long)SEQ * QKV, DH, QKV, 0,
         avb, (long long)XLEN * HD, DH, HD,
         XLEN, DH, SEQ, BS, NH, nullptr, nullptr, 0, 0, 0);

    // tmp = h + av @ out_w[i] ; h = LN1(tmp)
    gemm(avb, 0, 0, HD, ow_i, 0, 0, DM, 0, tmp, 0, 0, DM,
         BS * XLEN, DM, HD, 1, 1, nullptr, hbuf, DM, 0, 0);
    ln_kernel<<<BS * XLEN, 256, 0, stream>>>(tmp, ln1_g + (long long)i * DM,
                                             ln1_b + (long long)i * DM, hbuf);

    // ff = relu(h @ ff1_w + b1) ; tmp = h + ff @ ff2_w + b2 ; h = LN2(tmp)
    gemm(hbuf, 0, 0, DM, f1w_i, 0, 0, DI, 0, ffb, 0, 0, DI,
         BS * XLEN, DI, DM, 1, 1, f1b_i, nullptr, 0, /*relu*/1, 0);
    gemm(ffb, 0, 0, DI, f2w_i, 0, 0, DM, 0, tmp, 0, 0, DM,
         BS * XLEN, DM, DI, 1, 1, f2b_i, hbuf, DM, 0, 0);
    ln_kernel<<<BS * XLEN, 256, 0, stream>>>(tmp, ln2_g + (long long)i * DM,
                                             ln2_b + (long long)i * DM, hbuf);
  }

  hipMemcpyAsync(d_out, hbuf, (size_t)BS * XLEN * DM * sizeof(float),
                 hipMemcpyDeviceToDevice, stream);
}
